// VectorQuantizer_75849122447890
// MI455X (gfx1250) — matrix operand order repaired and checked by reading, not compiled, not run on hardware
//
#include <hip/hip_runtime.h>

// ---------------- Problem constants ----------------
#define NUM_EMB 1024
#define DIM     256
#define BATCH   32
#define HW      1024               // 32*32
#define NVEC    32768              // BATCH*HW
#define COMMIT  0.25f

// ---------------- Vector types ----------------
typedef __attribute__((ext_vector_type(16))) __bf16         v16bf;
typedef __attribute__((ext_vector_type(8)))  float          v8f;
typedef __attribute__((ext_vector_type(8)))  unsigned short u16x8;
typedef __attribute__((ext_vector_type(16))) unsigned short u16x16;

// ---------------- bf16 hi/lo split helpers ----------------
__device__ __forceinline__ unsigned short f32_to_bf16_rn(float f) {
  unsigned int u = __builtin_bit_cast(unsigned int, f);
  unsigned int r = u + 0x7FFFu + ((u >> 16) & 1u);   // round-to-nearest-even
  return (unsigned short)(r >> 16);
}
__device__ __forceinline__ float bf16_to_f32(unsigned short h) {
  unsigned int u = ((unsigned int)h) << 16;
  return __builtin_bit_cast(float, u);
}

// ---------------- Fragment loaders ----------------
// A-matrix 16x32 bf16 (ISA 7.12.2): lane r=lane&15 holds row M=r; hi=lane>>4.
// Lane's 16 elems: K = hi*8 + {0..7} then 16 + hi*8 + {0..7}  -> two 16B chunks.
__device__ __forceinline__ v16bf load_a_frag(const unsigned short* base /* row + kk + hi*8 */) {
  const u16x8* p = reinterpret_cast<const u16x8*>(base);
  u16x8 a = p[0];
  u16x8 b = p[2];                      // +16 ushorts
  u16x16 cat = __builtin_shufflevector(a, b, 0,1,2,3,4,5,6,7,8,9,10,11,12,13,14,15);
  return __builtin_bit_cast(v16bf, cat);
}
// B-matrix 32x16 bf16: lane holds col N=lane&15; K = hi*8 + {0..7} then 16 + hi*8 + {0..7}
// -> two 16B chunks per lane.
__device__ __forceinline__ v16bf load_b_frag(const unsigned short* base /* coderow + kk + hi*8 */) {
  const u16x8* p = reinterpret_cast<const u16x8*>(base);
  u16x8 a = p[0];
  u16x8 b = p[2];
  u16x16 cat = __builtin_shufflevector(a, b, 0,1,2,3,4,5,6,7,8,9,10,11,12,13,14,15);
  return __builtin_bit_cast(v16bf, cat);
}

// ---------------- Kernel 0: zero accumulators ----------------
__global__ void vq_init(unsigned int* __restrict__ counts, float* __restrict__ loss_acc) {
  int t = threadIdx.x;
  for (int j = 0; j < 4; ++j) counts[t * 4 + j] = 0u;
  if (t == 0) loss_acc[0] = 0.0f;
}

// ---------------- Kernel 1: normalize codebook rows -> bf16 hi/lo ----------------
__global__ void vq_codes_prep(const float* __restrict__ w,
                              unsigned short* __restrict__ w_hi,
                              unsigned short* __restrict__ w_lo) {
  int wave = threadIdx.x >> 5;
  int lane = threadIdx.x & 31;
  int row  = blockIdx.x * 8 + wave;
  const float* rp = w + (size_t)row * DIM;
  float v[8];
  float ss = 0.0f;
  #pragma unroll
  for (int j = 0; j < 8; ++j) { v[j] = rp[lane + 32 * j]; ss += v[j] * v[j]; }
  #pragma unroll
  for (int m = 1; m < 32; m <<= 1) ss += __shfl_xor(ss, m, 32);
  float inv = 1.0f / fmaxf(sqrtf(ss), 1e-12f);
  #pragma unroll
  for (int j = 0; j < 8; ++j) {
    float f = v[j] * inv;
    unsigned short hb = f32_to_bf16_rn(f);
    unsigned short lb = f32_to_bf16_rn(f - bf16_to_f32(hb));
    w_hi[(size_t)row * DIM + lane + 32 * j] = hb;
    w_lo[(size_t)row * DIM + lane + 32 * j] = lb;
  }
}

// ---------------- Kernel 2: BCHW -> [N,C] normalize + bf16 hi/lo split ----------------
// One block = (b, 32 consecutive hw). LDS-transposed with stride 257 (64 banks -> conflict-free).
__global__ void vq_flat_prep(const float* __restrict__ x,
                             unsigned short* __restrict__ f_hi,
                             unsigned short* __restrict__ f_lo) {
  __shared__ float tile[32 * 257];
  __shared__ float invn[32];
  int b   = blockIdx.x >> 5;
  int hw0 = (blockIdx.x & 31) << 5;
  int t = threadIdx.x;
  int lane = t & 31, wave = t >> 5;

  const float* xp = x + (size_t)b * DIM * HW + hw0 + lane;
  #pragma unroll 4
  for (int j = 0; j < 32; ++j) {
    int c = wave * 32 + j;
    tile[lane * 257 + c] = xp[(size_t)c * HW];        // coalesced reads
  }
  __syncthreads();

  int row = t >> 3, g = t & 7;                         // 8 threads per row
  float ss = 0.0f;
  #pragma unroll 4
  for (int j = 0; j < 32; ++j) {
    float v = tile[row * 257 + g * 32 + j];
    ss += v * v;
  }
  ss += __shfl_xor(ss, 1, 32);
  ss += __shfl_xor(ss, 2, 32);
  ss += __shfl_xor(ss, 4, 32);
  if (g == 0) invn[row] = 1.0f / fmaxf(sqrtf(ss), 1e-12f);
  __syncthreads();

  float inv = invn[row];
  size_t n = (size_t)b * HW + hw0 + row;
  #pragma unroll 4
  for (int j = 0; j < 32; ++j) {
    int c = g * 32 + j;
    float f = tile[row * 257 + c] * inv;
    unsigned short hb = f32_to_bf16_rn(f);
    unsigned short lb = f32_to_bf16_rn(f - bf16_to_f32(hb));
    f_hi[n * DIM + c] = hb;
    f_lo[n * DIM + c] = lb;
  }
}

// ---------------- Kernel 3: WMMA GEMM (hi/lo split) + row argmax ----------------
// Block tile: 64 rows x 1024 codes (4x less codebook L2 traffic than a 16-row tile).
// 8 waves arranged as 4 row-groups x 2 col-groups; each wave: 16 rows x 128 cols
// per chunk (8 accumulator tiles), looping over 4 col-chunks of 256 in increasing
// column order so strict '>' merging preserves jnp.argmin first-index tie-breaking.
__global__ void vq_gemm_argmax(const unsigned short* __restrict__ f_hi,
                               const unsigned short* __restrict__ f_lo,
                               const unsigned short* __restrict__ w_hi,
                               const unsigned short* __restrict__ w_lo,
                               int* __restrict__ out_idx) {
  __shared__ float sbv[8][16];
  __shared__ int   sbi[8][16];

  const int t      = threadIdx.x;
  const int wave   = t >> 5;
  const int lane   = t & 31;
  const int r      = lane & 15;
  const int hi     = lane >> 4;
  const int rowgrp = wave >> 1;          // 0..3
  const int colgrp = wave & 1;           // 0..1
  const int m_base = blockIdx.x * 64 + rowgrp * 16;

  const unsigned short* arow_hi = f_hi + (size_t)(m_base + r) * DIM + (hi << 3);
  const unsigned short* arow_lo = f_lo + (size_t)(m_base + r) * DIM + (hi << 3);

  // running per-row best (valid on all lanes after butterfly; row = e + hi*8)
  float rbv[8];
  int   rbc[8];
  #pragma unroll
  for (int e = 0; e < 8; ++e) { rbv[e] = -3.0e38f; rbc[e] = 0; }

  for (int nc = 0; nc < 4; ++nc) {
    const int colbase = nc * 256 + colgrp * 128;

    v8f acc[8];
    v8f z = {0.f, 0.f, 0.f, 0.f, 0.f, 0.f, 0.f, 0.f};
    #pragma unroll
    for (int tt = 0; tt < 8; ++tt) acc[tt] = z;

    for (int kk = 0; kk < DIM; kk += 32) {
      v16bf ah = load_a_frag(arow_hi + kk);
      v16bf al = load_a_frag(arow_lo + kk);
      #pragma unroll
      for (int tt = 0; tt < 8; ++tt) {
        const size_t crow = (size_t)(colbase + tt * 16 + r) * DIM + kk + (hi << 3);
        v16bf bh = load_b_frag(w_hi + crow);
        v16bf bl = load_b_frag(w_lo + crow);
        acc[tt] = __builtin_amdgcn_wmma_f32_16x16x32_bf16(false, ah, false, bh, (short)0, acc[tt], false, false);
        acc[tt] = __builtin_amdgcn_wmma_f32_16x16x32_bf16(false, ah, false, bl, (short)0, acc[tt], false, false);
        acc[tt] = __builtin_amdgcn_wmma_f32_16x16x32_bf16(false, al, false, bh, (short)0, acc[tt], false, false);
      }
    }

    // C layout: VGPR e holds row M=e (lanes 0-15) / M=e+8 (lanes 16-31); col = (lane&15)+16*tile.
    #pragma unroll
    for (int e = 0; e < 8; ++e) {
      float bv = -3.0e38f;
      int   bc = 0;
      #pragma unroll
      for (int tt = 0; tt < 8; ++tt) {
        float v = acc[tt][e];
        int   c = colbase + tt * 16 + r;
        if (v > bv) { bv = v; bc = c; }                // strict > keeps lowest col
      }
      // butterfly over each 16-lane half (masks <16 keep halves separate on wave32)
      #pragma unroll
      for (int m = 1; m < 16; m <<= 1) {
        float ov = __shfl_xor(bv, m, 32);
        int   oc = __shfl_xor(bc, m, 32);
        if (ov > bv || (ov == bv && oc < bc)) { bv = ov; bc = oc; }
      }
      // merge into running best (chunks visit increasing columns -> strict >)
      if (bv > rbv[e] || (bv == rbv[e] && bc < rbc[e])) { rbv[e] = bv; rbc[e] = bc; }
    }
  }

  #pragma unroll
  for (int e = 0; e < 8; ++e) {
    if (r == 0) {                                      // lanes 0 and 16
      int row = e + hi * 8;
      sbv[wave][row] = rbv[e];
      sbi[wave][row] = rbc[e];
    }
  }
  __syncthreads();

  if (t < 64) {                                        // one thread per block row
    int rg  = t >> 4;                                  // row-group
    int row = t & 15;
    float bv = sbv[2 * rg + 0][row];                   // colgrp 0 (lower cols) first
    int   bc = sbi[2 * rg + 0][row];
    float v1 = sbv[2 * rg + 1][row];
    int   c1 = sbi[2 * rg + 1][row];
    if (v1 > bv || (v1 == bv && c1 < bc)) { bv = v1; bc = c1; }
    out_idx[blockIdx.x * 64 + t] = bc;
  }
}

// ---------------- Kernel 4: gather quantized (BCHW) + loss partials ----------------
__global__ void vq_gather_loss(const float* __restrict__ x,
                               const float* __restrict__ w,
                               const int* __restrict__ idx,
                               float* __restrict__ q_out,
                               float* __restrict__ loss_acc) {
  __shared__ float red[8];
  size_t g = (size_t)blockIdx.x * 256 + threadIdx.x;   // BCHW linear index
  int hw = (int)(g & 1023);
  int c  = (int)((g >> 10) & 255);
  int b  = (int)(g >> 18);
  int n  = b * HW + hw;
  float q   = w[(size_t)idx[n] * DIM + c];             // codebook resident in L2
  float xin = x[g];
  q_out[g] = q;                                        // straight-through value == gather
  float d = q - xin;
  float s = d * d;
  #pragma unroll
  for (int m = 1; m < 32; m <<= 1) s += __shfl_xor(s, m, 32);
  int lane = threadIdx.x & 31, wv = threadIdx.x >> 5;
  if (lane == 0) red[wv] = s;
  __syncthreads();
  if (threadIdx.x == 0) {
    float tot = 0.f;
    #pragma unroll
    for (int i = 0; i < 8; ++i) tot += red[i];
    atomicAdd(loss_acc, tot);
  }
}

// ---------------- Kernel 5: one-hot encodings (zero-fill + set) + histogram ----------------
__global__ void vq_encodings(const int* __restrict__ idx,
                             float* __restrict__ enc,
                             unsigned int* __restrict__ counts) {
  int n  = blockIdx.x;
  int k0 = idx[n];
  int t  = threadIdx.x;
  int cbase = t * 4;
  float2 z0 = {0.f, 0.f}, z1 = {0.f, 0.f};
  if (k0 >= cbase && k0 < cbase + 4) {
    int off = k0 - cbase;
    if      (off == 0) z0.x = 1.f;
    else if (off == 1) z0.y = 1.f;
    else if (off == 2) z1.x = 1.f;
    else               z1.y = 1.f;
  }
  float2* p = reinterpret_cast<float2*>(enc + (size_t)n * NUM_EMB + cbase); // 8B aligned
  p[0] = z0;
  p[1] = z1;
  if (t == 0) atomicAdd(&counts[k0], 1u);
}

// ---------------- Kernel 6: loss + perplexity finalize ----------------
__global__ void vq_finalize(const unsigned int* __restrict__ counts,
                            const float* __restrict__ loss_acc,
                            float* __restrict__ out_loss,
                            float* __restrict__ out_perp) {
  __shared__ float red[8];
  int t = threadIdx.x;
  float s = 0.0f;
  #pragma unroll
  for (int j = 0; j < 4; ++j) {
    float p = (float)counts[t * 4 + j] * (1.0f / (float)NVEC);
    s += p * logf(p + 1e-10f);                         // p==0 contributes exactly 0
  }
  #pragma unroll
  for (int m = 1; m < 32; m <<= 1) s += __shfl_xor(s, m, 32);
  int lane = t & 31, wv = t >> 5;
  if (lane == 0) red[wv] = s;
  __syncthreads();
  if (t == 0) {
    float tot = 0.f;
    #pragma unroll
    for (int i = 0; i < 8; ++i) tot += red[i];
    out_loss[0] = loss_acc[0] * (COMMIT / ((float)NVEC * (float)DIM));
    out_perp[0] = expf(-tot);
  }
}

// ---------------- Host launcher ----------------
extern "C" void kernel_launch(void* const* d_in, const int* in_sizes, int n_in,
                              void* d_out, int out_size, void* d_ws, size_t ws_size,
                              hipStream_t stream) {
  (void)in_sizes; (void)n_in; (void)out_size; (void)ws_size;
  const float* x    = (const float*)d_in[0];   // [32,256,32,32]
  const float* embw = (const float*)d_in[1];   // [1024,256]
  float* out = (float*)d_out;

  // d_out layout (flat, return order): quantized [8388608] | loss [1] | perp [1] | encodings [33554432]
  float* q_out    = out;
  float* out_loss = out + 8388608;
  float* out_perp = out + 8388609;
  float* enc      = out + 8388610;

  // workspace layout
  char* ws = (char*)d_ws;
  unsigned short* f_hi = (unsigned short*)(ws);                       // 16 MB
  unsigned short* f_lo = (unsigned short*)(ws + 16777216);            // 16 MB
  unsigned short* w_hi = (unsigned short*)(ws + 33554432);            // 512 KB
  unsigned short* w_lo = (unsigned short*)(ws + 34078720);            // 512 KB
  int*            idx  = (int*)           (ws + 34603008);            // 128 KB
  unsigned int*   cnt  = (unsigned int*)  (ws + 34734080);            // 4 KB
  float*          lacc = (float*)         (ws + 34738176);            // 4 B

  vq_init        <<<dim3(1),     dim3(256), 0, stream>>>(cnt, lacc);
  vq_codes_prep  <<<dim3(128),   dim3(256), 0, stream>>>(embw, w_hi, w_lo);
  vq_flat_prep   <<<dim3(1024),  dim3(256), 0, stream>>>(x, f_hi, f_lo);
  vq_gemm_argmax <<<dim3(512),   dim3(256), 0, stream>>>(f_hi, f_lo, w_hi, w_lo, idx);
  vq_gather_loss <<<dim3(32768), dim3(256), 0, stream>>>(x, embw, idx, q_out, lacc);
  vq_encodings   <<<dim3(32768), dim3(256), 0, stream>>>(idx, enc, cnt);
  vq_finalize    <<<dim3(1),     dim3(256), 0, stream>>>(cnt, lacc, out_loss, out_perp);
}
